// BasicSubGraphLearner_62629213110676
// MI455X (gfx1250) — matrix operand order repaired and checked by reading, not compiled, not run on hardware
//
#include <hip/hip_runtime.h>
#include <hip/hip_bf16.h>

typedef __attribute__((ext_vector_type(16))) _Float16 v16h;
typedef __attribute__((ext_vector_type(8)))  float    v8f;

#define N_SEL   8192
#define N_FEAT  256
#define K_SUB   256
#define N_PERS  4
#define E_FULL  262144
#define E_RAW   262144
#define N_TOTAL 8192
#define EPSILON 0.5f
#define LAMB1   0.5f

// workspace layout (bytes)
#define WS_NSCORE  0u                 // 256 f32
#define WS_NORMS   4096u              // 4*8192 f32 = 131072 B
#define WS_BFRAG   (4096u + 131072u)  // 8*32*16 f16 = 8192 B
#define WS_BITMASK (1u << 20)         // N_SEL*N_SEL bits = 8 MB

// ---------------------------------------------------------------- zero
__global__ void zero_kernel(float4* __restrict__ out4, uint4* __restrict__ bm4,
                            long n_out4, long n_bm4) {
  long idx = (long)blockIdx.x * blockDim.x + threadIdx.x;
  long stride = (long)gridDim.x * blockDim.x;
  float4 zf = {0.f, 0.f, 0.f, 0.f};
  for (long i = idx; i < n_out4; i += stride) out4[i] = zf;
  uint4 zu = {0u, 0u, 0u, 0u};
  for (long i = idx; i < n_bm4; i += stride) bm4[i] = zu;
}

// ---------------------------------------------------------------- score normalization
__global__ void score_kernel(const float* __restrict__ score,
                             const int* __restrict__ belong,
                             float* __restrict__ nscore) {
  __shared__ float ssum[K_SUB];
  int t = threadIdx.x;
  ssum[t] = 0.f;
  __syncthreads();
  atomicAdd(&ssum[belong[t]], score[t]);
  __syncthreads();
  nscore[t] = score[t] / ssum[belong[t]];
}

// ---------------------------------------------------------------- per-row per-perspective norms
__global__ void norms_kernel(const float* __restrict__ x,
                             const float* __restrict__ w,
                             float* __restrict__ norms) {
  int t = blockIdx.x * blockDim.x + threadIdx.x;   // [0, 4*N_SEL)
  int p = t & 3;
  int r = t >> 2;
  const float* xr = x + (long)r * N_FEAT;
  const float* wp = w + p * N_FEAT;
  float acc = 0.f;
  for (int d = 0; d < N_FEAT; d += 4) {
    float4 xv = *(const float4*)(xr + d);
    float4 wv = *(const float4*)(wp + d);
    float a0 = xv.x * wv.x, a1 = xv.y * wv.y, a2 = xv.z * wv.z, a3 = xv.w * wv.w;
    acc += a0 * a0 + a1 * a1 + a2 * a2 + a3 * a3;
  }
  norms[p * N_SEL + r] = sqrtf(acc) + 1e-12f;
}

// ---------------------------------------------------------------- pack B fragment: w_p[k]^2 in ISA 32x16 f16 layout
// bfrag[(c8*32 + lane)*16 + c] : component c of lane's v16h for k-chunk c8.
// 16-bit B (KxN): VGPR r, lane<16 -> K=2r,2r+1 (r<4) / 8+2r,9+2r (r>=4) N=lane ; lane>=16 -> K+8 N=lane-16
// => component c maps to K = 8*(lane>=16) + c + (c>=8 ? 8 : 0), N = lane & 15.
__global__ void bfrag_kernel(const float* __restrict__ w, _Float16* __restrict__ bfrag) {
  int t = threadIdx.x;       // 256 = 8 chunks * 32 lanes
  int c8 = t >> 5;
  int lane = t & 31;
  int n = lane & 15;
  int khi = (lane >= 16) ? 8 : 0;
  for (int c = 0; c < 16; ++c) {
    int k = c8 * 32 + khi + c + (c >= 8 ? 8 : 0);
    float v = 0.f;
    if (n < N_PERS) { float wv = w[n * N_FEAT + k]; v = wv * wv; }
    bfrag[(t << 4) + c] = (_Float16)v;
  }
}

// ---------------------------------------------------------------- main: 1 wave = 16 edges, WMMA over K=256
__global__ void __launch_bounds__(256) edge_wmma_kernel(
    const float* __restrict__ x,
    const int* __restrict__ fei,       // [2][E_FULL]
    const int* __restrict__ sbatch,
    const int* __restrict__ smap,
    const float* __restrict__ nscore,
    const float* __restrict__ norms,   // [4][N_SEL]
    const v16h* __restrict__ bfrag,    // [8*32] fragments
    unsigned int* __restrict__ bitmask,
    float* __restrict__ out) {
  __shared__ float dtile[8][16][16];
  int lane = threadIdx.x & 31;
  int wave = threadIdx.x >> 5;
  int tile = blockIdx.x * 8 + wave;
  int m  = lane & 15;      // edge within tile == A row == D row
  int hv = lane >> 4;      // lane half selects K sub-ranges
  int e  = tile * 16 + m;
  int i  = fei[e];
  int j  = fei[E_FULL + e];

  // Dedup: reference's mask.set(1.0) collapses duplicate (i,j) edges.
  bool isNew = false;
  if (hv == 0) {
    unsigned int cell = (unsigned int)i * (unsigned int)N_SEL + (unsigned int)j;
    unsigned int old = atomicOr(&bitmask[cell >> 5], 1u << (cell & 31));
    isNew = ((old & (1u << (cell & 31))) == 0u);
  }

  const float* xi = x + (long)i * N_FEAT;
  const float* xj = x + (long)j * N_FEAT;
  int o1 = hv * 8;   // 16-bit A 16x32 layout: lane<16 K={0-7,16-23}, lane>=16 K={8-15,24-31}

  v8f acc = {0.f, 0.f, 0.f, 0.f, 0.f, 0.f, 0.f, 0.f};
#pragma unroll
  for (int c8 = 0; c8 < 8; ++c8) {
    int kb = c8 * 32 + o1;
    float4 xi0 = *(const float4*)(xi + kb);
    float4 xi1 = *(const float4*)(xi + kb + 4);
    float4 xi2 = *(const float4*)(xi + kb + 16);
    float4 xi3 = *(const float4*)(xi + kb + 20);
    float4 xj0 = *(const float4*)(xj + kb);
    float4 xj1 = *(const float4*)(xj + kb + 4);
    float4 xj2 = *(const float4*)(xj + kb + 16);
    float4 xj3 = *(const float4*)(xj + kb + 20);
    v16h a;
    a[0]  = (_Float16)(xi0.x * xj0.x);
    a[1]  = (_Float16)(xi0.y * xj0.y);
    a[2]  = (_Float16)(xi0.z * xj0.z);
    a[3]  = (_Float16)(xi0.w * xj0.w);
    a[4]  = (_Float16)(xi1.x * xj1.x);
    a[5]  = (_Float16)(xi1.y * xj1.y);
    a[6]  = (_Float16)(xi1.z * xj1.z);
    a[7]  = (_Float16)(xi1.w * xj1.w);
    a[8]  = (_Float16)(xi2.x * xj2.x);
    a[9]  = (_Float16)(xi2.y * xj2.y);
    a[10] = (_Float16)(xi2.z * xj2.z);
    a[11] = (_Float16)(xi2.w * xj2.w);
    a[12] = (_Float16)(xi3.x * xj3.x);
    a[13] = (_Float16)(xi3.y * xj3.y);
    a[14] = (_Float16)(xi3.z * xj3.z);
    a[15] = (_Float16)(xi3.w * xj3.w);
    v16h b = bfrag[c8 * 32 + lane];
    // D[m,p] += sum_k x_i[k]*x_j[k]*w_p[k]^2
    acc = __builtin_amdgcn_wmma_f32_16x16x32_f16(false, a, false, b,
                                                 (short)0, acc, false, false);
  }

  // D layout: VGPR v -> M = v + 8*hv, N = lane&15. Bounce through LDS to get row-per-edge.
#pragma unroll
  for (int v = 0; v < 8; ++v) dtile[wave][v + 8 * hv][m] = acc[v];
  __syncthreads();

  if (hv == 0) {
    float s = 0.f;
#pragma unroll
    for (int p = 0; p < N_PERS; ++p) {
      float ni = norms[p * N_SEL + i];
      float nj = norms[p * N_SEL + j];
      s += dtile[wave][m][p] / (ni * nj);
    }
    s *= (1.0f / (float)N_PERS);
    if (isNew && (i != j) && (s > EPSILON)) {
      float wgt = s * nscore[sbatch[i]] * LAMB1;
      atomicAdd(&out[(long)smap[i] * N_TOTAL + smap[j]], wgt);
    }
  }
}

// ---------------------------------------------------------------- raw edges: duplicates DO accumulate
__global__ void raw_kernel(const int* __restrict__ rei, float* __restrict__ out) {
  int e = blockIdx.x * blockDim.x + threadIdx.x;
  if (e < E_RAW) {
    atomicAdd(&out[(long)rei[e] * N_TOTAL + rei[E_RAW + e]], 1.0f - LAMB1);
  }
}

// ---------------------------------------------------------------- launch
extern "C" void kernel_launch(void* const* d_in, const int* in_sizes, int n_in,
                              void* d_out, int out_size, void* d_ws, size_t ws_size,
                              hipStream_t stream) {
  const float* x       = (const float*)d_in[0];
  const float* mweight = (const float*)d_in[1];
  const int*   sbatch  = (const int*)d_in[2];
  const int*   smap    = (const int*)d_in[3];
  const int*   sbelong = (const int*)d_in[4];
  const float* sscore  = (const float*)d_in[5];
  const int*   fei     = (const int*)d_in[6];
  const int*   rei     = (const int*)d_in[7];
  float* out = (float*)d_out;

  char* ws = (char*)d_ws;
  float*        nscore  = (float*)(ws + WS_NSCORE);
  float*        norms   = (float*)(ws + WS_NORMS);
  _Float16*     bfrag   = (_Float16*)(ws + WS_BFRAG);
  unsigned int* bitmask = (unsigned int*)(ws + WS_BITMASK);

  long n_out4 = (long)N_TOTAL * N_TOTAL / 4;            // 16,777,216 float4
  long n_bm4  = (long)N_SEL * N_SEL / 32 / 4;           // 524,288 uint4
  zero_kernel<<<4096, 256, 0, stream>>>((float4*)out, (uint4*)bitmask, n_out4, n_bm4);

  score_kernel<<<1, K_SUB, 0, stream>>>(sscore, sbelong, nscore);
  norms_kernel<<<(N_SEL * N_PERS) / 256, 256, 0, stream>>>(x, mweight, norms);
  bfrag_kernel<<<1, 256, 0, stream>>>(mweight, bfrag);

  int tiles = E_FULL / 16;                               // 16384 waves
  edge_wmma_kernel<<<tiles / 8, 256, 0, stream>>>(
      x, fei, sbatch, smap, nscore, norms, (const v16h*)bfrag, bitmask, out);

  raw_kernel<<<E_RAW / 256, 256, 0, stream>>>(rei, out);
}
